// NeuralEmotionAnalyzer_3204045603450
// MI455X (gfx1250) — matrix operand order repaired and checked by reading, not compiled, not run on hardware
//
#include <hip/hip_runtime.h>

// ---------------------------------------------------------------------------
// NeuralEmotionAnalyzer on MI455X (gfx1250): bf16 WMMA GEMM pipeline.
// All heavy GEMMs use v_wmma_f32_16x16x32_bf16 (f32 accumulate).
// A-tiles staged with global_load_async_to_lds_b128 (ASYNCcnt path);
// double-buffered LDS pipeline overlaps global traffic with WMMA compute.
// Dead computation (q/k projections; softmax over length-1 key) is skipped.
// ---------------------------------------------------------------------------

typedef __bf16 bf16;
typedef __attribute__((ext_vector_type(16))) __bf16 v16bf;
typedef __attribute__((ext_vector_type(8)))  float  v8f;
typedef __attribute__((ext_vector_type(4)))  __bf16 bf16x4;

union FragU { uint4 u[2]; v16bf v; };

__device__ __forceinline__ float geluf(float x) {
    return 0.5f * x * (1.0f + erff(x * 0.70710678118654752f));
}
__device__ __forceinline__ float sigmf(float x) {
    return 1.0f / (1.0f + __expf(-x));
}

// ---------------------------------------------------------------------------
// Generic tiled GEMM: C[M,N] = A[M,K](bf16) * W[N,K](f32, cvt->bf16)^T + bias
// Block tile 128x128, BK=64, 256 threads = 8 waves, wave tile 64x32 (4x2 WMMA).
// Double-buffered LDS (2 x 32KB).
// ---------------------------------------------------------------------------
template <typename OutT, bool DO_GELU>
__global__ __launch_bounds__(256) void gemm_bf16_wmma(
    const bf16* __restrict__ A, int lda,
    const float* __restrict__ W,
    const float* __restrict__ bias,
    OutT* __restrict__ C, int ldc, int K)
{
    __shared__ bf16 sA[2 * 128 * 64];
    __shared__ bf16 sB[2 * 128 * 64];

    const int tid  = threadIdx.x;
    const int m0   = blockIdx.y * 128;
    const int n0   = blockIdx.x * 128;
    const int lane = tid & 31;
    const int wave = tid >> 5;
    const int wm   = wave & 1;   // 2 wave-rows -> 64 M each
    const int wn   = wave >> 1;  // 4 wave-cols -> 32 N each
    const int hf   = lane >> 4;  // half-wave id
    const int l15  = lane & 15;

    // Per-thread staging geometry (constant across K loop).
    const int aRow = tid >> 3;            // + i*32 rows, 4 chunks of 8 bf16
    const int aKc  = (tid & 7) << 3;
    unsigned aLds[4];
    const bf16* aGp[4];
    #pragma unroll
    for (int i = 0; i < 4; ++i) {
        int row  = aRow + i * 32;
        aLds[i]  = (unsigned)(unsigned long long)(const void*)&sA[row * 64 + aKc];
        aGp[i]   = A + (long)(m0 + row) * lda + aKc;
    }
    const int wRow = tid >> 4;            // + i*16 rows, 8 chunks of 4 f32
    const int wKc  = (tid & 15) << 2;

    v8f acc[4][2] = {};
    float4 wf[8];

    auto stageA = [&](int k0, int buf) {   // async global -> LDS (ASYNCcnt)
        #pragma unroll
        for (int i = 0; i < 4; ++i) {
            const bf16* ga = aGp[i] + k0;
            unsigned la = aLds[i] + buf * 16384;
            asm volatile("global_load_async_to_lds_b128 %0, %1, off"
                         :: "v"(la), "v"(ga) : "memory");
        }
    };
    auto loadW = [&](int k0) {             // batched f32 global loads
        #pragma unroll
        for (int i = 0; i < 8; ++i) {
            int row = wRow + i * 16;
            wf[i] = *reinterpret_cast<const float4*>(W + (long)(n0 + row) * K + k0 + wKc);
        }
    };
    auto storeW = [&](int buf) {           // cvt f32 -> bf16, store to LDS
        #pragma unroll
        for (int i = 0; i < 8; ++i) {
            int row = wRow + i * 16;
            bf16x4 h;
            h.x = (bf16)wf[i].x; h.y = (bf16)wf[i].y;
            h.z = (bf16)wf[i].z; h.w = (bf16)wf[i].w;
            *reinterpret_cast<bf16x4*>(&sB[buf * 8192 + row * 64 + wKc]) = h;
        }
    };
    auto compute = [&](int buf) {
        const bf16* pA = sA + buf * 8192;
        const bf16* pB = sB + buf * 8192;
        #pragma unroll
        for (int ks = 0; ks < 64; ks += 32) {
            v16bf fa[4], fb[2];
            // A frag (16x32 bf16): lanes 0-15 hold M=l15, K {0..7,16..23};
            //                      lanes 16-31 hold K {8..15,24..31}.
            #pragma unroll
            for (int tm = 0; tm < 4; ++tm) {
                int m  = wm * 64 + tm * 16 + l15;
                int kb = ks + hf * 8;
                FragU u;
                u.u[0] = *reinterpret_cast<const uint4*>(&pA[m * 64 + kb]);
                u.u[1] = *reinterpret_cast<const uint4*>(&pA[m * 64 + kb + 16]);
                fa[tm] = u.v;
            }
            // B frag (32x16 bf16): lanes 0-15 hold col N=l15 K {0..7,16..23}; lanes 16-31 K {8..15,24..31}.
            #pragma unroll
            for (int tn = 0; tn < 2; ++tn) {
                int n  = wn * 32 + tn * 16 + l15;
                int kb = ks + hf * 8;
                FragU u;
                u.u[0] = *reinterpret_cast<const uint4*>(&pB[n * 64 + kb]);
                u.u[1] = *reinterpret_cast<const uint4*>(&pB[n * 64 + kb + 16]);
                fb[tn] = u.v;
            }
            #pragma unroll
            for (int tm = 0; tm < 4; ++tm)
                #pragma unroll
                for (int tn = 0; tn < 2; ++tn)
                    acc[tm][tn] = __builtin_amdgcn_wmma_f32_16x16x32_bf16(
                        false, fa[tm], false, fb[tn], (short)0, acc[tm][tn], false, false);
        }
    };

    // ---- pipelined main loop (double buffered) ----
    const int nkt = K >> 6;
    stageA(0, 0);
    loadW(0);
    storeW(0);
    asm volatile("s_wait_asynccnt 0" ::: "memory");
    __syncthreads();
    for (int kt = 0; kt < nkt; ++kt) {
        const int buf = kt & 1;
        if (kt + 1 < nkt) {
            stageA((kt + 1) << 6, buf ^ 1);   // prefetch next tile (async)
            loadW((kt + 1) << 6);             // next W tile in flight
        }
        compute(buf);
        if (kt + 1 < nkt) {
            storeW(buf ^ 1);
            asm volatile("s_wait_asynccnt 0" ::: "memory");
            __syncthreads();
        }
    }

    // Epilogue: C/D layout — lanes 0-15: VGPR r = M r, N=lane; lanes 16-31: M 8+r.
    #pragma unroll
    for (int tm = 0; tm < 4; ++tm) {
        #pragma unroll
        for (int tn = 0; tn < 2; ++tn) {
            int gn   = n0 + wn * 32 + tn * 16 + l15;
            float bv = bias[gn];
            #pragma unroll
            for (int r = 0; r < 8; ++r) {
                long m  = (long)m0 + wm * 64 + tm * 16 + hf * 8 + r;
                float v = acc[tm][tn][r] + bv;
                if (DO_GELU) v = geluf(v);
                C[m * (long)ldc + gn] = (OutT)v;
            }
        }
    }
}

// ---------------------------------------------------------------------------
// Row LayerNorm + exact GELU, f32 -> bf16.  One 256-thread block per row.
// Params gamma/beta have P rows (P=5 for cultural detectors), row r uses r%P.
// ---------------------------------------------------------------------------
__global__ __launch_bounds__(256) void ln_gelu_bf16(
    const float* __restrict__ src, bf16* __restrict__ dst,
    const float* __restrict__ g, const float* __restrict__ b, int C, int P)
{
    __shared__ float red[256];
    const long r = blockIdx.x;
    const float* x = src + r * C;
    float s = 0.f, s2 = 0.f;
    for (int j = threadIdx.x; j < C; j += 256) { float v = x[j]; s += v; s2 += v * v; }
    red[threadIdx.x] = s; __syncthreads();
    for (int o = 128; o > 0; o >>= 1) { if (threadIdx.x < o) red[threadIdx.x] += red[threadIdx.x + o]; __syncthreads(); }
    float S = red[0]; __syncthreads();
    red[threadIdx.x] = s2; __syncthreads();
    for (int o = 128; o > 0; o >>= 1) { if (threadIdx.x < o) red[threadIdx.x] += red[threadIdx.x + o]; __syncthreads(); }
    float S2 = red[0];
    float mean = S / C;
    float var  = S2 / C - mean * mean;
    float rstd = rsqrtf(var + 1e-5f);
    const float* gp = g + (long)(r % P) * C;
    const float* bp = b + (long)(r % P) * C;
    bf16* y = dst + r * C;
    for (int j = threadIdx.x; j < C; j += 256) {
        float v = (x[j] - mean) * rstd * gp[j] + bp[j];
        y[j] = (bf16)geluf(v);
    }
}

// LSTM cell activation (seq_len=1, zero state): h = sig(o)*tanh(sig(i)*tanh(g))
__global__ void lstm_act(const float* __restrict__ g4, bf16* outB, float* outF,
                         int colOff, long n)
{
    long idx = (long)blockIdx.x * blockDim.x + threadIdx.x;
    if (idx >= n) return;
    long b = idx >> 10;
    int  j = (int)(idx & 1023);
    long base = b * 4096;
    float i  = g4[base + j];
    float gg = g4[base + 2048 + j];
    float o  = g4[base + 3072 + j];
    float c  = sigmf(i) * tanhf(gg);
    float h  = sigmf(o) * tanhf(c);
    if (outB) outB[b * 2048 + colOff + j] = (bf16)h;
    if (outF) outF[b * 2048 + colOff + j] = h;
}

// gate = softmax(gelu(gh) @ w2^T + b2); one thread per batch row (N=8 tiny).
__global__ void gate_kernel(const float* __restrict__ gh, const float* __restrict__ w2,
                            const float* __restrict__ b2, float* __restrict__ gate, int Bt)
{
    int b = blockIdx.x * blockDim.x + threadIdx.x;
    if (b >= Bt) return;
    float acc[8];
    #pragma unroll
    for (int e = 0; e < 8; ++e) acc[e] = b2[e];
    const float* xr = gh + (long)b * 768;
    for (int j = 0; j < 768; ++j) {
        float ga = geluf(xr[j]);
        #pragma unroll
        for (int e = 0; e < 8; ++e) acc[e] += ga * w2[e * 768 + j];
    }
    float mx = acc[0];
    #pragma unroll
    for (int e = 1; e < 8; ++e) mx = fmaxf(mx, acc[e]);
    float sum = 0.f;
    #pragma unroll
    for (int e = 0; e < 8; ++e) { acc[e] = __expf(acc[e] - mx); sum += acc[e]; }
    float inv = 1.0f / sum;
    #pragma unroll
    for (int e = 0; e < 8; ++e) gate[(long)b * 8 + e] = acc[e] * inv;
}

// cultural[b,q] = mean_k( c3[b,k,:] . w4[k,q,:] + b4[k,q] ), N=3 tiny.
__global__ void cul_final(const float* __restrict__ c3, const float* __restrict__ w4,
                          const float* __restrict__ b4, float* __restrict__ out, int Bt)
{
    int b = blockIdx.x * blockDim.x + threadIdx.x;
    if (b >= Bt) return;
    #pragma unroll
    for (int q = 0; q < 3; ++q) {
        float s = 0.f;
        for (int k = 0; k < 5; ++k) {
            float d = b4[k * 3 + q];
            const float* xr = c3 + (long)b * 1920 + k * 384;
            const float* wr = w4 + (long)(k * 3 + q) * 384;
            for (int o = 0; o < 384; ++o) d += xr[o] * wr[o];
            s += d;
        }
        out[(long)b * 3 + q] = s * 0.2f;
    }
}

// moe[b,p] = sum_e gate[b,e] * ( e2[b,e,:] . w3[e,p,:] + b3[e,p] ), N=7 tiny.
__global__ void moe_final(const float* __restrict__ e2, const float* __restrict__ gate,
                          const float* __restrict__ w3, const float* __restrict__ b3,
                          float* __restrict__ out, int Bt)
{
    long gid = (long)blockIdx.x * blockDim.x + threadIdx.x;
    long b = gid >> 3;
    int  p = (int)(gid & 7);
    if (b >= Bt || p >= 7) return;
    float acc = 0.f;
    for (int e = 0; e < 8; ++e) {
        float d = b3[e * 7 + p];
        const float* xr = e2 + b * 3072 + e * 384;
        const float* wr = w3 + (long)(e * 7 + p) * 384;
        for (int o = 0; o < 384; ++o) d += xr[o] * wr[o];
        acc += gate[b * 8 + e] * d;
    }
    out[b * 7 + p] = acc;
}

__global__ void f32_to_bf16(const float* __restrict__ s, bf16* __restrict__ d, long n)
{
    long i = (long)blockIdx.x * blockDim.x + threadIdx.x;
    if (i < n) d[i] = (bf16)s[i];
}

__global__ void zero_f32(float* __restrict__ p, long n)
{
    long i = (long)blockIdx.x * blockDim.x + threadIdx.x;
    if (i < n) p[i] = 0.0f;
}

// ---------------------------------------------------------------------------
extern "C" void kernel_launch(void* const* d_in, const int* in_sizes, int n_in,
                              void* d_out, int out_size, void* d_ws, size_t ws_size,
                              hipStream_t stream)
{
    constexpr int B = 8192;
    const float* x        = (const float*)d_in[0];
    const float* enc_w1   = (const float*)d_in[1];
    const float* enc_b1   = (const float*)d_in[2];
    const float* enc_ln1g = (const float*)d_in[3];
    const float* enc_ln1b = (const float*)d_in[4];
    const float* enc_w2   = (const float*)d_in[5];
    const float* enc_b2   = (const float*)d_in[6];
    const float* enc_ln2g = (const float*)d_in[7];
    const float* enc_ln2b = (const float*)d_in[8];
    const float* enc_w3   = (const float*)d_in[9];
    const float* enc_b3   = (const float*)d_in[10];
    // d_in[11..14] = wq/bq/wk/bk -> dead (softmax over single key == 1)
    const float* attn_wv  = (const float*)d_in[15];
    const float* attn_bv  = (const float*)d_in[16];
    const float* attn_wo  = (const float*)d_in[17];
    const float* attn_bo  = (const float*)d_in[18];
    const float* fus_w1   = (const float*)d_in[19];
    const float* fus_b1   = (const float*)d_in[20];
    const float* fus_lng  = (const float*)d_in[21];
    const float* fus_lnb  = (const float*)d_in[22];
    const float* fus_w2   = (const float*)d_in[23];
    const float* fus_b2   = (const float*)d_in[24];
    const float* lstm_w0  = (const float*)d_in[25];
    const float* lstm_b0  = (const float*)d_in[26];
    const float* lstm_w12 = (const float*)d_in[27];
    const float* lstm_b12 = (const float*)d_in[28];
    const float* cul_w1   = (const float*)d_in[29];
    const float* cul_b1   = (const float*)d_in[30];
    const float* cul_lng  = (const float*)d_in[31];
    const float* cul_lnb  = (const float*)d_in[32];
    const float* cul_w2   = (const float*)d_in[33];
    const float* cul_b2   = (const float*)d_in[34];
    const float* cul_w3   = (const float*)d_in[35];
    const float* cul_b3   = (const float*)d_in[36];
    const float* cul_w4   = (const float*)d_in[37];
    const float* cul_b4   = (const float*)d_in[38];
    const float* gate_w1  = (const float*)d_in[39];
    const float* gate_b1  = (const float*)d_in[40];
    const float* gate_w2  = (const float*)d_in[41];
    const float* gate_b2  = (const float*)d_in[42];
    const float* moe_w1   = (const float*)d_in[43];
    const float* moe_b1   = (const float*)d_in[44];
    const float* moe_w2   = (const float*)d_in[45];
    const float* moe_b2   = (const float*)d_in[46];
    const float* moe_w3   = (const float*)d_in[47];
    const float* moe_b3   = (const float*)d_in[48];

    // Output layout (f32, concatenated flat): moe | cultural | temporal | bio
    float* out      = (float*)d_out;
    float* out_moe  = out;                       // [B,7]
    float* out_cul  = out + (long)B * 7;         // [B,3]
    float* out_temp = out + (long)B * 10;        // [B,2048]
    float* out_bio  = out + (long)B * 10 + (long)B * 2048;  // [B,512]

    // Workspace layout
    char* ws = (char*)d_ws;
    float* F     = (float*)ws;                                      // B*5120 f32
    bf16*  BA    = (bf16*)(ws + (size_t)B * 5120 * 4);              // B*6144 bf16
    bf16*  BB    = (bf16*)((char*)BA + (size_t)B * 6144 * 2);       // B*3840 bf16
    bf16*  FUSED = (bf16*)((char*)BB + (size_t)B * 3840 * 2);       // B*1536 bf16
    float* GATE  = (float*)((char*)FUSED + (size_t)B * 1536 * 2);   // B*8 f32

    const dim3 blk(256);
    auto gemmF = [&](const bf16* A, int lda, const float* W, const float* bias,
                     float* C, int ldc, int N, int K) {
        dim3 g(N / 128, B / 128);
        gemm_bf16_wmma<float, false><<<g, blk, 0, stream>>>(A, lda, W, bias, C, ldc, K);
    };
    auto gemmB = [&](const bf16* A, int lda, const float* W, const float* bias,
                     bf16* C, int ldc, int N, int K) {
        dim3 g(N / 128, B / 128);
        gemm_bf16_wmma<bf16, false><<<g, blk, 0, stream>>>(A, lda, W, bias, C, ldc, K);
    };
    auto gemmBG = [&](const bf16* A, int lda, const float* W, const float* bias,
                      bf16* C, int ldc, int N, int K) {
        dim3 g(N / 128, B / 128);
        gemm_bf16_wmma<bf16, true><<<g, blk, 0, stream>>>(A, lda, W, bias, C, ldc, K);
    };

    // ---- encoder ----
    {   long n = (long)B * 896;
        f32_to_bf16<<<dim3((n + 255) / 256), blk, 0, stream>>>(x, BB, n); }
    gemmF(BB, 896, enc_w1, enc_b1, F, 2048, 2048, 896);
    ln_gelu_bf16<<<dim3(B), blk, 0, stream>>>(F, BA, enc_ln1g, enc_ln1b, 2048, 1);
    gemmF(BA, 2048, enc_w2, enc_b2, F, 2048, 2048, 2048);
    ln_gelu_bf16<<<dim3(B), blk, 0, stream>>>(F, BB, enc_ln2g, enc_ln2b, 2048, 1);
    gemmB(BB, 2048, enc_w3, enc_b3, BA, 1536, 1536, 2048);           // enc -> BA

    // ---- attention (softmax over single key == 1) ----
    gemmB(BA, 1536, attn_wv, attn_bv, BB, 1536, 1536, 1536);         // v   -> BB
    gemmB(BB, 1536, attn_wo, attn_bo, BA, 1536, 1536, 1536);         // attn-> BA

    // ---- fusion MLP ----
    gemmF(BA, 1536, fus_w1, fus_b1, F, 2048, 2048, 1536);
    ln_gelu_bf16<<<dim3(B), blk, 0, stream>>>(F, BB, fus_lng, fus_lnb, 2048, 1);
    gemmB(BB, 2048, fus_w2, fus_b2, FUSED, 1536, 1536, 2048);        // fused (persistent)

    // ---- 3-layer bidirectional LSTM (seq_len=1) ----
    const long nact = (long)B * 1024;
    const dim3 gact((nact + 255) / 256);
    for (int d = 0; d < 2; ++d) {
        gemmF(FUSED, 1536, lstm_w0 + (long)d * 4096 * 1536, lstm_b0 + (long)d * 4096,
              F, 4096, 4096, 1536);
        lstm_act<<<gact, blk, 0, stream>>>(F, BA, nullptr, d * 1024, nact);
    }
    for (int d = 0; d < 2; ++d) {
        gemmF(BA, 2048, lstm_w12 + (long)d * 4096 * 2048, lstm_b12 + (long)d * 4096,
              F, 4096, 4096, 2048);
        lstm_act<<<gact, blk, 0, stream>>>(F, BB, nullptr, d * 1024, nact);
    }
    for (int d = 0; d < 2; ++d) {
        gemmF(BB, 2048, lstm_w12 + (long)(2 + d) * 4096 * 2048, lstm_b12 + (long)(2 + d) * 4096,
              F, 4096, 4096, 2048);
        lstm_act<<<gact, blk, 0, stream>>>(F, nullptr, out_temp, d * 1024, nact);
    }

    // ---- 5 cultural detectors ----
    gemmF(FUSED, 1536, cul_w1, cul_b1, F, 5120, 5120, 1536);         // c1 raw
    ln_gelu_bf16<<<dim3(B * 5), blk, 0, stream>>>(F, BA, cul_lng, cul_lnb, 1024, 5);
    for (int k = 0; k < 5; ++k)
        gemmBG(BA + k * 1024, 5120, cul_w2 + (long)k * 768 * 1024, cul_b2 + k * 768,
               BB + k * 768, 3840, 768, 1024);                        // c2 = gelu(...)
    for (int k = 0; k < 5; ++k)
        gemmF(BB + k * 768, 3840, cul_w3 + (long)k * 384 * 768, cul_b3 + k * 384,
              F + k * 384, 1920, 384, 768);                           // c3
    cul_final<<<dim3((B + 255) / 256), blk, 0, stream>>>(F, cul_w4, cul_b4, out_cul, B);

    // ---- MoE gate ----
    gemmF(FUSED, 1536, gate_w1, gate_b1, F, 768, 768, 1536);
    gate_kernel<<<dim3((B + 255) / 256), blk, 0, stream>>>(F, gate_w2, gate_b2, GATE, B);

    // ---- 8 experts ----
    gemmBG(FUSED, 1536, moe_w1, moe_b1, BA, 6144, 6144, 1536);       // e1 = gelu(...)
    for (int e = 0; e < 8; ++e)
        gemmF(BA + e * 768, 6144, moe_w2 + (long)e * 384 * 768, moe_b2 + e * 384,
              F + e * 384, 3072, 384, 768);                           // e2
    {   long n = (long)B * 8;
        moe_final<<<dim3((n + 255) / 256), blk, 0, stream>>>(F, GATE, moe_w3, moe_b3, out_moe, B); }

    // ---- bio = zeros ----
    {   long n = (long)B * 512;
        zero_f32<<<dim3((n + 255) / 256), blk, 0, stream>>>(out_bio, n); }
}
